// mmTEM_20375324852566
// MI455X (gfx1250) — matrix operand order repaired and checked by reading, not compiled, not run on hardware
//
#include <hip/hip_runtime.h>
#include <hip/hip_bf16.h>
#include <math.h>

// ---------------------------------------------------------------------------
// MI455X (gfx1250) implementation notes:
//  * All dense layers run through v_wmma_f32_16x16x32_f16 (wave32, fp32 accum).
//  * Weights are pre-converted ONCE to f16 column-major with K padded to a
//    multiple of 32 -> GEMM inner loop: only 128-bit loads, no bounds checks,
//    no f32->f16 conversion on the B side.
//  * Tile width is templated: 16x128 per wave (8 WMMA / A-fragment) for the
//    dominant 1024x16384 GEMM, 16x64 for narrow layers (occupancy).
//  * Output is dominated by ~540MB of per-sample gradient writes -> grad_write
//    is a pure streaming-store kernel (HBM-bound at 23.3 TB/s, ~23us floor).
//    Grad regions are 4B-aligned (out+1) so coalesced b32 stores are used.
//  * Losses use fixed-order reductions => deterministic across graph replays.
// ---------------------------------------------------------------------------

typedef __attribute__((ext_vector_type(16))) _Float16 v16h;
typedef __attribute__((ext_vector_type(8)))  _Float16 v8h;
typedef __attribute__((ext_vector_type(8)))  float    v8f;
typedef __attribute__((ext_vector_type(4)))  float    v4f;

#define BS 1024   // B*S tokens; every GEMM here has M == 1024 (grid.y == 64)

// ---------------------------------------------------------------------------
// GEMM: C[1024 x N] = act(A[1024 x K] (row stride lda) @ W + bias)
// W given as Wt = f16 column-major [N][K], K % 32 == 0 (zero-padded).
// One wave computes a 16 x (NT*16) tile: NT accumulators share one A fragment.
// Fragment layouts per CDNA5 ISA 7.12.2:
//   A 16x32 f16 : lane m=l%16, khalf=l/16 ; elems = A[m][k0+8h .. +8) ++
//                 A[m][k0+16+8h .. +8)          -> four 16B loads
//   B 32x16 f16 : lane n=l%16, khalf=l/16 ; elems = Wt[n][k0+8h .. +8) ++
//                 Wt[n][k0+16+8h .. +8)          -> two 16B loads
//   C 16x16 f32 : lane n=l%16, mhalf=l/16 ; vgpr r: m = r + 8*mhalf
// ---------------------------------------------------------------------------
template <int NT>
__global__ __launch_bounds__(32) void gemm_wmma_f16(
    const float* __restrict__ A, int lda,
    const _Float16* __restrict__ Wt,
    const float* __restrict__ bias,
    float* __restrict__ C, int N, int K, int relu)
{
  const int lane = threadIdx.x;
  const int l16  = lane & 15;
  const int half = lane >> 4;
  const int m0   = blockIdx.y << 4;
  const int n0   = blockIdx.x * (NT * 16);

  const float*    __restrict__ Arow = A  + (size_t)(m0 + l16) * lda;
  const _Float16* __restrict__ Wc0  = Wt + (size_t)(n0 + l16) * K + (half << 3);

  const v8f zero = {};
  v8f acc[NT];
#pragma unroll
  for (int t = 0; t < NT; ++t) acc[t] = zero;

  for (int k0 = 0; k0 < K; k0 += 32) {
    __builtin_prefetch(Arow + k0 + 32, 0, 1);          // global_prefetch_b8
    const v4f a0 = *(const v4f*)(Arow + k0 + (half << 3));
    const v4f a1 = *(const v4f*)(Arow + k0 + (half << 3) + 4);
    const v4f a2 = *(const v4f*)(Arow + k0 + 16 + (half << 3));
    const v4f a3 = *(const v4f*)(Arow + k0 + 16 + (half << 3) + 4);
    v16h a;
#pragma unroll
    for (int e = 0; e < 4; ++e) {
      a[e]      = (_Float16)a0[e];
      a[e + 4]  = (_Float16)a1[e];
      a[e + 8]  = (_Float16)a2[e];
      a[e + 12] = (_Float16)a3[e];
    }
#pragma unroll
    for (int t = 0; t < NT; ++t) {
      const _Float16* __restrict__ Wc = Wc0 + (size_t)(t << 4) * K + k0;
      const v8h b0 = *(const v8h*)(Wc);
      const v8h b1 = *(const v8h*)(Wc + 16);
      v16h b;
#pragma unroll
      for (int e = 0; e < 8; ++e) { b[e] = b0[e]; b[e + 8] = b1[e]; }
      acc[t] = __builtin_amdgcn_wmma_f32_16x16x32_f16(
          false, a, false, b, (short)0, acc[t], false, false);
    }
  }
#pragma unroll
  for (int t = 0; t < NT; ++t) {
    const int col = n0 + (t << 4) + l16;
    const float bv = bias ? bias[col] : 0.f;
#pragma unroll
    for (int r = 0; r < 8; ++r) {
      const int m = m0 + r + (half << 3);
      float v = acc[t][r] + bv;
      if (relu) v = fmaxf(v, 0.f);
      C[(size_t)m * N + col] = v;
    }
  }
}

// ---- weight prep: f32 row-major [R x C] -> f16 column-major [C][Kp], zero pad
__global__ void wconv_col_kernel(const float* __restrict__ src,
                                 _Float16* __restrict__ dst, int R, int C, int Kp)
{
  const int idx = blockIdx.x * 256 + threadIdx.x;
  if (idx >= C * Kp) return;
  const int c = idx / Kp, k = idx - c * Kp;
  dst[idx] = (k < R) ? (_Float16)src[(size_t)k * C + c] : (_Float16)0.f;
}

// ---- plain f32 -> f16 copy (row-major copy of W == column-major of W^T)
__global__ void wconv_copy_kernel(const float* __restrict__ src,
                                  _Float16* __restrict__ dst, int n)
{
  const int idx = blockIdx.x * 256 + threadIdx.x;
  if (idx < n) dst[idx] = (_Float16)src[idx];
}

// ---------------------------------------------------------------------------
// Sequential path-integration scan: h <- l2norm(relu(h @ T[s])) per step.
// 1024 threads/block: 8 K-slices per output element, LDS tree reductions.
// This is the latency-critical kernel (128 strictly sequential 64KB matvecs).
// ---------------------------------------------------------------------------
__global__ __launch_bounds__(1024) void scan_kernel(
    const float* __restrict__ trans, const float* __restrict__ init_h,
    float* __restrict__ structural)
{
  __shared__ float h[128];
  __shared__ float ysh[128];
  __shared__ float red[1024];
  const int tid = threadIdx.x;
  const int j = tid & 127, p = tid >> 7;   // p in 0..7: K-slice
  const int b = blockIdx.x;

  if (tid < 128) { const float hv = init_h[tid]; ysh[tid] = hv; red[tid] = hv * hv; }
  __syncthreads();
  if (tid < 64) red[tid] += red[tid + 64]; __syncthreads();
  if (tid < 32) red[tid] += red[tid + 32]; __syncthreads();
  if (tid < 16) red[tid] += red[tid + 16]; __syncthreads();
  if (tid <  8) red[tid] += red[tid +  8]; __syncthreads();
  if (tid <  4) red[tid] += red[tid +  4]; __syncthreads();
  if (tid <  2) red[tid] += red[tid +  2]; __syncthreads();
  if (tid <  1) red[tid] += red[tid +  1]; __syncthreads();
  if (tid < 128) h[tid] = ysh[tid] / fmaxf(sqrtf(red[0]), 1e-12f);
  __syncthreads();

  for (int s = 0; s < 128; ++s) {
    const float* __restrict__ T =
        trans + ((size_t)(b * 128 + s)) * 16384 + (size_t)(p << 4) * 128 + j;
    float yp = 0.f;
#pragma unroll
    for (int q = 0; q < 16; ++q) yp = fmaf(h[(p << 4) + q], T[q * 128], yp);
    red[tid] = yp; __syncthreads();
    if (tid < 512) red[tid] += red[tid + 512]; __syncthreads();
    if (tid < 256) red[tid] += red[tid + 256]; __syncthreads();
    if (tid < 128) ysh[tid] = fmaxf(red[tid] + red[tid + 128], 0.f);
    __syncthreads();
    if (tid < 128) red[tid] = ysh[tid] * ysh[tid];
    __syncthreads();
    if (tid < 64) red[tid] += red[tid + 64]; __syncthreads();
    if (tid < 32) red[tid] += red[tid + 32]; __syncthreads();
    if (tid < 16) red[tid] += red[tid + 16]; __syncthreads();
    if (tid <  8) red[tid] += red[tid +  8]; __syncthreads();
    if (tid <  4) red[tid] += red[tid +  4]; __syncthreads();
    if (tid <  2) red[tid] += red[tid +  2]; __syncthreads();
    if (tid <  1) red[tid] += red[tid +  1]; __syncthreads();
    if (tid < 128) {
      const float hn = ysh[tid] / fmaxf(sqrtf(red[0]), 1e-12f);
      h[tid] = hn;
      structural[((size_t)(b * 128 + s)) * 128 + tid] = hn;
    }
    __syncthreads();
  }
}

// dst[rows x (ca+cb)] = [a | b], null source => zero-fill
__global__ void concat2_kernel(float* __restrict__ dst,
                               const float* __restrict__ a, int sa, int ca,
                               const float* __restrict__ b, int sb, int cb, int rows)
{
  const int cols = ca + cb;
  const int idx = blockIdx.x * blockDim.x + threadIdx.x;
  if (idx >= rows * cols) return;
  const int r = idx / cols, c = idx - r * cols;
  float v;
  if (c < ca) v = a ? a[(size_t)r * sa + c]        : 0.f;
  else        v = b ? b[(size_t)r * sb + (c - ca)] : 0.f;
  dst[idx] = v;
}

// dst[rows x 288] = [corr_s | dec_gen_s | sse | 0-pad] (K padded 257 -> 288)
__global__ void concat_sin_kernel(float* __restrict__ dst,
                                  const float* __restrict__ cs, int scs,
                                  const float* __restrict__ dgs, int sdgs,
                                  const float* __restrict__ sse, int rows)
{
  const int idx = blockIdx.x * blockDim.x + threadIdx.x;
  if (idx >= rows * 288) return;
  const int r = idx / 288, c = idx - r * 288;
  float v;
  if      (c < 128)  v = cs [(size_t)r * scs  + c];
  else if (c < 256)  v = dgs[(size_t)r * sdgs + (c - 128)];
  else if (c == 256) v = sse[r];
  else               v = 0.f;
  dst[idx] = v;
}

__global__ void sse_kernel(const float* __restrict__ ce, int sce,
                           const float* __restrict__ enc,
                           float* __restrict__ sse, int rows)
{
  const int t = blockIdx.x * blockDim.x + threadIdx.x;
  if (t >= rows) return;
  float s = 0.f;
  for (int c = 0; c < 128; ++c) {
    const float d = ce[(size_t)t * sce + c] - enc[(size_t)t * 128 + c];
    s = fmaf(d, d, s);
  }
  sse[t] = s;
}

// inf_s = dgs + (cs - dgs) * ratio * pv
__global__ void infs_kernel(const float* __restrict__ dgs, int sd,
                            const float* __restrict__ cs, int sc,
                            const float* __restrict__ pv,
                            const float* __restrict__ ratio,
                            float* __restrict__ infs, int rows)
{
  const int idx = blockIdx.x * blockDim.x + threadIdx.x;
  if (idx >= rows * 128) return;
  const int r = idx >> 7, c = idx & 127;
  const float g = dgs[(size_t)r * sd + c];
  const float k = cs [(size_t)r * sc + c];
  infs[idx] = g + (k - g) * ratio[0] * pv[idx];
}

// dy = (2/128) * (y - v)
__global__ void dy_kernel(const float* __restrict__ y, const float* __restrict__ v,
                          float* __restrict__ dy, int n)
{
  const int i = blockIdx.x * blockDim.x + threadIdx.x;
  if (i < n) dy[i] = (2.0f / 128.0f) * (y[i] - v[i]);
}

// x *= (h > 0)   (relu backward mask; h is post-activation)
__global__ void mask_kernel(float* __restrict__ x, const float* __restrict__ h, int n)
{
  const int i = blockIdx.x * blockDim.x + threadIdx.x;
  if (i < n) x[i] = (h[i] > 0.f) ? x[i] : 0.f;
}

// deterministic MSE partials: partials[block] = scale * block_sum((a-b)^2) / n
__global__ __launch_bounds__(256) void mse_partial_kernel(
    const float* __restrict__ a, int sa, const float* __restrict__ b, int sb,
    int rows, int cols, float scale, float* __restrict__ partials)
{
  __shared__ float red[256];
  const int tid = threadIdx.x;
  const int n = rows * cols;
  float s = 0.f;
  for (int idx = blockIdx.x * 256 + tid; idx < n; idx += 64 * 256) {
    const int r = idx / cols, c = idx - r * cols;
    const float d = a[(size_t)r * sa + c] - b[(size_t)r * sb + c];
    s = fmaf(d, d, s);
  }
  red[tid] = s; __syncthreads();
  for (int off = 128; off; off >>= 1) { if (tid < off) red[tid] += red[tid + off]; __syncthreads(); }
  if (tid == 0) partials[blockIdx.x] = scale * red[0] / (float)n;
}

// fixed-order sum of 4 x 64 partials -> deterministic total loss
__global__ void finalize_kernel(const float* __restrict__ p, float* __restrict__ out)
{
  if (threadIdx.x == 0 && blockIdx.x == 0) {
    float s = 0.f;
    for (int i = 0; i < 256; ++i) s += p[i];
    out[0] = s;
  }
}

// ---------------------------------------------------------------------------
// Streaming per-token gradient writes (~540MB of stores; HBM-bound).
// Grad regions sit at out+1 (4B alignment) -> coalesced b32 stores.
// ---------------------------------------------------------------------------
__global__ __launch_bounds__(256) void grad_write_kernel(
    const float* __restrict__ keys, const float* __restrict__ h1,
    const float* __restrict__ h2,   const float* __restrict__ dy,
    const float* __restrict__ dh1,  const float* __restrict__ dh2,
    float* __restrict__ gW1, float* __restrict__ gb1,
    float* __restrict__ gW2, float* __restrict__ gb2,
    float* __restrict__ gW3, float* __restrict__ gb3)
{
  const int t = blockIdx.x, tid = threadIdx.x;
  __shared__ float k_[128], h1_[256], h2_[256], dy_[128], dh1_[256], dh2_[256];
  if (tid < 128) { k_[tid] = keys[(size_t)t * 128 + tid]; dy_[tid] = dy[(size_t)t * 128 + tid]; }
  h1_[tid]  = h1 [(size_t)t * 256 + tid];
  h2_[tid]  = h2 [(size_t)t * 256 + tid];
  dh1_[tid] = dh1[(size_t)t * 256 + tid];
  dh2_[tid] = dh2[(size_t)t * 256 + tid];
  __syncthreads();

  float* w1 = gW1 + (size_t)t * 32768;
  float* w2 = gW2 + (size_t)t * 65536;
  float* w3 = gW3 + (size_t)t * 32768;
  for (int idx = tid; idx < 32768; idx += 256) { const int i = idx >> 8, j = idx & 255; w1[idx] = k_[i]  * dh1_[j]; }
  for (int idx = tid; idx < 65536; idx += 256) { const int i = idx >> 8, j = idx & 255; w2[idx] = h1_[i] * dh2_[j]; }
  for (int idx = tid; idx < 32768; idx += 256) { const int i = idx >> 7, j = idx & 127; w3[idx] = h2_[i] * dy_[j];  }
  gb1[(size_t)t * 256 + tid] = dh1_[tid];
  gb2[(size_t)t * 256 + tid] = dh2_[tid];
  if (tid < 128) gb3[(size_t)t * 128 + tid] = dy_[tid];
}

// ---------------------------------------------------------------------------
// Host orchestration
// ---------------------------------------------------------------------------
extern "C" void kernel_launch(void* const* d_in, const int* in_sizes, int n_in,
                              void* d_out, int out_size, void* d_ws, size_t ws_size,
                              hipStream_t stream)
{
  (void)in_sizes; (void)n_in; (void)out_size; (void)ws_size;
  const float* sensory = (const float*)d_in[0];
  const float* actions = (const float*)d_in[1];
  const float* init_h  = (const float*)d_in[2];
  const float* tW1 = (const float*)d_in[3];  const float* tb1 = (const float*)d_in[4];
  const float* tW2 = (const float*)d_in[5];  const float* tb2 = (const float*)d_in[6];
  const float* tW3 = (const float*)d_in[7];  const float* tb3 = (const float*)d_in[8];
  const float* enc_w = (const float*)d_in[9];  const float* enc_b = (const float*)d_in[10];
  const float* dec_w = (const float*)d_in[11]; const float* dec_b = (const float*)d_in[12];
  const float* q_w = (const float*)d_in[13];
  const float* k_w = (const float*)d_in[14];
  const float* v_w = (const float*)d_in[15];
  const float* mW1 = (const float*)d_in[16]; const float* mb1 = (const float*)d_in[17];
  const float* mW2 = (const float*)d_in[18]; const float* mb2 = (const float*)d_in[19];
  const float* mW3 = (const float*)d_in[20]; const float* mb3 = (const float*)d_in[21];
  const float* oW1 = (const float*)d_in[22]; const float* ob1 = (const float*)d_in[23];
  const float* oW2 = (const float*)d_in[24]; const float* ob2 = (const float*)d_in[25];
  const float* oW3 = (const float*)d_in[26]; const float* ob3 = (const float*)d_in[27];
  const float* sW1 = (const float*)d_in[28]; const float* sb1 = (const float*)d_in[29];
  const float* sW2 = (const float*)d_in[30]; const float* sb2 = (const float*)d_in[31];
  const float* sW3 = (const float*)d_in[32]; const float* sb3 = (const float*)d_in[33];
  const float* ratio = (const float*)d_in[34];

  // ---- workspace layout (float offsets) ----
  float* ws = (float*)d_ws;
  constexpr size_t WS_TRANS = 0;                         // 1024*16384
  constexpr size_t WS_T1    = WS_TRANS + 16777216;
  constexpr size_t WS_T2    = WS_T1    + 131072;
  constexpr size_t WS_STRUC = WS_T2    + 131072;
  constexpr size_t WS_ENC   = WS_STRUC + 131072;
  constexpr size_t WS_JOINT = WS_ENC   + 131072;
  constexpr size_t WS_Q     = WS_JOINT + 262144;
  constexpr size_t WS_R1    = WS_Q     + 131072;
  constexpr size_t WS_R2    = WS_R1    + 262144;
  constexpr size_t WS_R3    = WS_R2    + 262144;
  constexpr size_t WS_O1    = WS_R3    + 131072;
  constexpr size_t WS_O2    = WS_O1    + 262144;
  constexpr size_t WS_OUTA  = WS_O2    + 262144;
  constexpr size_t WS_OUTB  = WS_OUTA  + 262144;
  constexpr size_t WS_OUTC  = WS_OUTB  + 262144;
  constexpr size_t WS_OUTD  = WS_OUTC  + 262144;
  constexpr size_t WS_RECON = WS_OUTD  + 262144;         // 1024*512 (reused)
  constexpr size_t WS_SSE   = WS_RECON + 524288;
  constexpr size_t WS_SIN   = WS_SSE   + 1024;           // 1024*288 (padded)
  constexpr size_t WS_S1    = WS_SIN   + 294912;
  constexpr size_t WS_S2    = WS_S1    + 262144;
  constexpr size_t WS_PV    = WS_S2    + 262144;
  constexpr size_t WS_INFS  = WS_PV    + 131072;
  constexpr size_t WS_STORE = WS_INFS  + 131072;
  constexpr size_t WS_KEYS  = WS_STORE + 262144;
  constexpr size_t WS_VALS  = WS_KEYS  + 131072;
  constexpr size_t WS_H1    = WS_VALS  + 131072;
  constexpr size_t WS_H2    = WS_H1    + 262144;
  constexpr size_t WS_Y     = WS_H2    + 262144;
  constexpr size_t WS_DY    = WS_Y     + 131072;
  constexpr size_t WS_DH2   = WS_DY    + 131072;
  constexpr size_t WS_DH1   = WS_DH2   + 262144;
  constexpr size_t WS_LOSS  = WS_DH1   + 262144;         // 4 x 64 partials
  constexpr size_t WS_F16   = WS_LOSS  + 256;            // f16 weight region

  _Float16* wh = (_Float16*)(ws + WS_F16);
  // f16 weight offsets (halves), each [N][Kp] column-major, Kp % 32 == 0
  constexpr size_t HT1  = 0;               // 128 x 32
  constexpr size_t HT2  = HT1  + 4096;     // 128 x 128
  constexpr size_t HT3  = HT2  + 16384;    // 16384 x 128
  constexpr size_t HENC = HT3  + 2097152;  // 128 x 512
  constexpr size_t HDEC = HENC + 65536;    // 512 x 128
  constexpr size_t HQ   = HDEC + 65536;    // 128 x 256
  constexpr size_t HK   = HQ   + 32768;
  constexpr size_t HV   = HK   + 32768;
  constexpr size_t HM1  = HV   + 32768;    // 256 x 128
  constexpr size_t HM2  = HM1  + 32768;    // 256 x 256
  constexpr size_t HM3  = HM2  + 65536;    // 128 x 256
  constexpr size_t HO1  = HM3  + 32768;
  constexpr size_t HO2  = HO1  + 32768;
  constexpr size_t HO3  = HO2  + 65536;
  constexpr size_t HS1  = HO3  + 65536;    // 256 x 288 (K 257 -> 288)
  constexpr size_t HS2  = HS1  + 73728;
  constexpr size_t HS3  = HS2  + 65536;
  constexpr size_t HM3T = HS3  + 32768;    // col-major of mW3^T == f16 copy of mW3
  constexpr size_t HM2T = HM3T + 32768;    // col-major of mW2^T == f16 copy of mW2

  auto wcol = [&](const float* src, _Float16* dst, int R, int C, int Kp) {
    const int n = C * Kp;
    wconv_col_kernel<<<(n + 255) / 256, 256, 0, stream>>>(src, dst, R, C, Kp);
  };
  // ---- weight prep (tiny, once per launch) ----
  wcol(tW1,   wh + HT1,  32,  128,   32);
  wcol(tW2,   wh + HT2,  128, 128,   128);
  wcol(tW3,   wh + HT3,  128, 16384, 128);
  wcol(enc_w, wh + HENC, 512, 128,   512);
  wcol(dec_w, wh + HDEC, 128, 512,   128);
  wcol(q_w,   wh + HQ,   256, 128,   256);
  wcol(k_w,   wh + HK,   256, 128,   256);
  wcol(v_w,   wh + HV,   256, 128,   256);
  wcol(mW1,   wh + HM1,  128, 256,   128);
  wcol(mW2,   wh + HM2,  256, 256,   256);
  wcol(mW3,   wh + HM3,  256, 128,   256);
  wcol(oW1,   wh + HO1,  128, 256,   128);
  wcol(oW2,   wh + HO2,  256, 256,   256);
  wcol(oW3,   wh + HO3,  256, 256,   256);
  wcol(sW1,   wh + HS1,  257, 256,   288);
  wcol(sW2,   wh + HS2,  256, 256,   256);
  wcol(sW3,   wh + HS3,  256, 128,   256);
  wconv_copy_kernel<<<128, 256, 0, stream>>>(mW3, wh + HM3T, 32768);
  wconv_copy_kernel<<<256, 256, 0, stream>>>(mW2, wh + HM2T, 65536);

  auto gemm = [&](const float* A, int lda, const _Float16* Wt, const float* b,
                  float* C, int N, int K, int relu) {
    if (N >= 1024) {           // dominant wide GEMM: 16x128 tile per wave
      dim3 grid(N / 128, 64);
      gemm_wmma_f16<8><<<grid, 32, 0, stream>>>(A, lda, Wt, b, C, N, K, relu);
    } else {                   // narrow layers: 16x64 tile, more blocks
      dim3 grid(N / 64, 64);
      gemm_wmma_f16<4><<<grid, 32, 0, stream>>>(A, lda, Wt, b, C, N, K, relu);
    }
  };
  auto retrieve = [&](const float* joint, float* out) {
    gemm(joint,        256, wh + HQ,  nullptr, ws + WS_Q,  128, 256, 0);
    gemm(ws + WS_Q,    128, wh + HM1, mb1,     ws + WS_R1, 256, 128, 1);
    gemm(ws + WS_R1,   256, wh + HM2, mb2,     ws + WS_R2, 256, 256, 1);
    gemm(ws + WS_R2,   256, wh + HM3, mb3,     ws + WS_R3, 128, 256, 0);
    gemm(ws + WS_R3,   128, wh + HO1, ob1,     ws + WS_O1, 256, 128, 1);
    gemm(ws + WS_O1,   256, wh + HO2, ob2,     ws + WS_O2, 256, 256, 1);
    gemm(ws + WS_O2,   256, wh + HO3, ob3,     out,        256, 256, 0);
  };

  // ---- 1. transitions MLP (dominant GEMM: 1024 x 128 -> 16384) ----
  gemm(actions,    32,  wh + HT1, tb1, ws + WS_T1,    128,   32,  1);
  gemm(ws + WS_T1, 128, wh + HT2, tb2, ws + WS_T2,    128,   128, 1);
  gemm(ws + WS_T2, 128, wh + HT3, tb3, ws + WS_TRANS, 16384, 128, 0);

  // ---- 2. sequential RNN scan -> structural ----
  scan_kernel<<<8, 1024, 0, stream>>>(ws + WS_TRANS, init_h, ws + WS_STRUC);

  // ---- 3. sensory encoder ----
  gemm(sensory, 512, wh + HENC, enc_b, ws + WS_ENC, 128, 512, 0);

  // ---- 4. generative branch: retrieve(structural, 0) ----
  concat2_kernel<<<1024, 256, 0, stream>>>(ws + WS_JOINT, ws + WS_STRUC, 128, 128,
                                           nullptr, 0, 128, BS);
  retrieve(ws + WS_JOINT, ws + WS_OUTA);
  gemm(ws + WS_OUTA + 128, 256, wh + HDEC, dec_b, ws + WS_RECON, 512, 128, 0);
  mse_partial_kernel<<<64, 256, 0, stream>>>(sensory, 512, ws + WS_RECON, 512,
                                             BS, 512, 1.f, ws + WS_LOSS + 0);

  // ---- 5. relational branch: retrieve(0, encoded) x2 (identical -> scale 2) ----
  concat2_kernel<<<1024, 256, 0, stream>>>(ws + WS_JOINT, nullptr, 0, 128,
                                           ws + WS_ENC, 128, 128, BS);
  retrieve(ws + WS_JOINT, ws + WS_OUTB);
  mse_partial_kernel<<<64, 256, 0, stream>>>(ws + WS_OUTB, 256, ws + WS_STRUC, 128,
                                             BS, 128, 2.f, ws + WS_LOSS + 64);

  // ---- 6. correction: retrieve(dec_gen_s, encoded) ----
  concat2_kernel<<<1024, 256, 0, stream>>>(ws + WS_JOINT, ws + WS_OUTA, 256, 128,
                                           ws + WS_ENC, 128, 128, BS);
  retrieve(ws + WS_JOINT, ws + WS_OUTC);
  sse_kernel<<<4, 256, 0, stream>>>(ws + WS_OUTC + 128, 256, ws + WS_ENC, ws + WS_SSE, BS);

  // ---- 7. pred_var MLP (K padded 257 -> 288) ----
  concat_sin_kernel<<<(BS * 288 + 255) / 256, 256, 0, stream>>>(
      ws + WS_SIN, ws + WS_OUTC, 256, ws + WS_OUTA, 256, ws + WS_SSE, BS);
  gemm(ws + WS_SIN, 288, wh + HS1, sb1, ws + WS_S1, 256, 288, 1);
  gemm(ws + WS_S1,  256, wh + HS2, sb2, ws + WS_S2, 256, 256, 1);
  gemm(ws + WS_S2,  256, wh + HS3, sb3, ws + WS_PV, 128, 256, 0);
  infs_kernel<<<512, 256, 0, stream>>>(ws + WS_OUTA, 256, ws + WS_OUTC, 256,
                                       ws + WS_PV, ratio, ws + WS_INFS, BS);
  mse_partial_kernel<<<64, 256, 0, stream>>>(ws + WS_OUTA, 256, ws + WS_INFS, 128,
                                             BS, 128, 1.f, ws + WS_LOSS + 128);

  // ---- 8. inference branch: retrieve(inf_s, 0) ----
  concat2_kernel<<<1024, 256, 0, stream>>>(ws + WS_JOINT, ws + WS_INFS, 128, 128,
                                           nullptr, 0, 128, BS);
  retrieve(ws + WS_JOINT, ws + WS_OUTD);
  gemm(ws + WS_OUTD + 128, 256, wh + HDEC, dec_b, ws + WS_RECON, 512, 128, 0);
  mse_partial_kernel<<<64, 256, 0, stream>>>(sensory, 512, ws + WS_RECON, 512,
                                             BS, 512, 1.f, ws + WS_LOSS + 192);

  // ---- 9. fast-weight memory grads ----
  concat2_kernel<<<1024, 256, 0, stream>>>(ws + WS_STORE, ws + WS_OUTD, 256, 128,
                                           ws + WS_ENC, 128, 128, BS);
  gemm(ws + WS_STORE, 256, wh + HK, nullptr, ws + WS_KEYS, 128, 256, 0);
  gemm(ws + WS_STORE, 256, wh + HV, nullptr, ws + WS_VALS, 128, 256, 0);
  // meta forward
  gemm(ws + WS_KEYS, 128, wh + HM1, mb1, ws + WS_H1, 256, 128, 1);
  gemm(ws + WS_H1,   256, wh + HM2, mb2, ws + WS_H2, 256, 256, 1);
  gemm(ws + WS_H2,   256, wh + HM3, mb3, ws + WS_Y,  128, 256, 0);
  // backward (GEMMs against W^T == f16 row-major copies)
  dy_kernel<<<512, 256, 0, stream>>>(ws + WS_Y, ws + WS_VALS, ws + WS_DY, BS * 128);
  gemm(ws + WS_DY,  128, wh + HM3T, nullptr, ws + WS_DH2, 256, 128, 0);
  mask_kernel<<<1024, 256, 0, stream>>>(ws + WS_DH2, ws + WS_H2, BS * 256);
  gemm(ws + WS_DH2, 256, wh + HM2T, nullptr, ws + WS_DH1, 256, 256, 0);
  mask_kernel<<<1024, 256, 0, stream>>>(ws + WS_DH1, ws + WS_H1, BS * 256);

  // ---- 10. outputs ----
  float* out = (float*)d_out;
  constexpr size_t OW1 = 1;
  constexpr size_t OB1 = OW1 + (size_t)BS * 32768;
  constexpr size_t OW2 = OB1 + (size_t)BS * 256;
  constexpr size_t OB2 = OW2 + (size_t)BS * 65536;
  constexpr size_t OW3 = OB2 + (size_t)BS * 256;
  constexpr size_t OB3 = OW3 + (size_t)BS * 32768;
  grad_write_kernel<<<BS, 256, 0, stream>>>(
      ws + WS_KEYS, ws + WS_H1, ws + WS_H2, ws + WS_DY, ws + WS_DH1, ws + WS_DH2,
      out + OW1, out + OB1, out + OW2, out + OB2, out + OW3, out + OB3);
  finalize_kernel<<<1, 32, 0, stream>>>(ws + WS_LOSS, out);
}
